// SparseAttention_89635967468052
// MI455X (gfx1250) — matrix operand order repaired and checked by reading, not compiled, not run on hardware
//
#include <hip/hip_runtime.h>
#include <stdint.h>

// ---------------------------------------------------------------- constants
#define BB 4
#define NN 4096
#define EE 1024
#define HH 16
#define DD 64
#define M_TOTAL (BB * NN)       // 16384 tokens / GEMM rows
#define W_ELEMS (EE * EE)       // elements per weight matrix (contiguous in ws)

// ---------------------------------------------------------------- types
typedef __bf16 bf16x16 __attribute__((ext_vector_type(16)));
typedef float  f32x8   __attribute__((ext_vector_type(8)));
typedef float  f32x4   __attribute__((ext_vector_type(4)));
typedef unsigned int   u32x4 __attribute__((ext_vector_type(4)));
typedef unsigned short u16x4 __attribute__((ext_vector_type(4)));

union FragAB {            // 32 bytes = 8 VGPRs: one WMMA A or B operand
    bf16x16 bf;
    u32x4   q[2];
};

static __device__ __forceinline__ unsigned short f32_to_bf16(float f) {
    unsigned int u = __float_as_uint(f);
    unsigned int lsb = (u >> 16) & 1u;
    u += 0x7fffu + lsb;               // round-to-nearest-even
    return (unsigned short)(u >> 16);
}

// ---------------------------------------------------------------- fp32 -> bf16
__global__ __launch_bounds__(256) void cvt_bf16_kernel(
        const float* __restrict__ src, unsigned short* __restrict__ dst, int n4) {
    int i = (blockIdx.x * blockDim.x + threadIdx.x);
    if (i < n4) {
        const f32x4 v = ((const f32x4*)src)[i];
        u16x4 o;
        o[0] = f32_to_bf16(v[0]);
        o[1] = f32_to_bf16(v[1]);
        o[2] = f32_to_bf16(v[2]);
        o[3] = f32_to_bf16(v[3]);
        ((u16x4*)dst)[i] = o;
    }
}

// ---------------------------------------------------------------- fused QKV GEMM
// y[m][n] = sum_k x[m][k] * W[n][k] + b[n]   (both operands K-contiguous)
// Wave tile: 16(M) x 64(N) for each of Q/K/V. Block = 8 waves covering 512 N cols.
// All fragment addresses are ONE per-lane base + compile-time immediate.
// B fragments: 3-buffer cyclic pipeline, loaded 2 fragments ahead (the (fi+2)%3
// pattern is iteration-invariant -> no loop-carried copies, no address selects).
// A fragments: double-buffered via two explicit K half-steps.
// grid = (M_TOTAL/16, EE/512)

// B-fragment element offset for fragment fi at K position kk
#define BOFF(fi, kk) ((size_t)((fi) >> 2) * W_ELEMS + (size_t)((fi) & 3) * 16 * EE + (kk))

#define LOADB(dst, off)                                         \
    do {                                                        \
        (dst).q[0] = *(const u32x4*)(bbase + (off));            \
        (dst).q[1] = *(const u32x4*)(bbase + (off) + 16);       \
    } while (0)

// 12 WMMAs at K position KK using A fragment AREG; preload 2 fragments ahead.
#define GEMM_STEP(KK, AREG)                                                   \
    do {                                                                      \
        _Pragma("unroll")                                                     \
        for (int fi = 0; fi < 12; ++fi) {                                     \
            const int tf = fi + 2;          /* preload target */              \
            if (tf < 12) LOADB(bb[tf % 3], BOFF(tf, (KK)));                   \
            else         LOADB(bb[tf % 3], BOFF(tf - 12, (KK) + 32));         \
            acc[fi] = __builtin_amdgcn_wmma_f32_16x16x32_bf16(                \
                false, (AREG).bf, false, bb[fi % 3].bf, (short)0,             \
                acc[fi], false, false);                                       \
        }                                                                     \
    } while (0)

__global__ __launch_bounds__(256) void qkv_gemm_kernel(
        const unsigned short* __restrict__ xb,
        const unsigned short* __restrict__ wbase,   // [3][EE][EE] q,k,v contiguous
        const float* __restrict__ bq,
        const float* __restrict__ bk,
        const float* __restrict__ bv,
        unsigned short* __restrict__ qo,     // [M_TOTAL][EE]  (row-major, h*64+d)
        unsigned short* __restrict__ ko,     // [M_TOTAL][EE]
        unsigned short* __restrict__ vto)    // [M_TOTAL][DD][HH] transposed per token
{
    const int lane  = threadIdx.x & 31;
    const int wave  = threadIdx.x >> 5;
    const int col   = lane & 15;     // tile column (N) / A row (M)
    const int hh    = lane >> 4;     // lane half
    const int mrow  = blockIdx.x * 16;
    const int nbase = blockIdx.y * 512 + wave * 64;

    const float* bias[3] = { bq, bk, bv };

    f32x8 acc[12];                    // fi = m*4 + t
    #pragma unroll
    for (int m = 0; m < 3; ++m) {
        #pragma unroll
        for (int t = 0; t < 4; ++t) {
            float b = bias[m][nbase + t * 16 + col];   // per-column bias, all rows
            acc[m * 4 + t] = f32x8{b, b, b, b, b, b, b, b};
        }
    }

    // single per-lane bases; everything else is an immediate offset
    const unsigned short* abase = xb + (size_t)(mrow + col) * EE + hh * 8;
    const unsigned short* bbase = wbase + (size_t)(nbase + col) * EE + hh * 8;

    FragAB a0, a1, bb[3];
    a0.q[0] = *(const u32x4*)(abase);
    a0.q[1] = *(const u32x4*)(abase + 16);
    LOADB(bb[0], BOFF(0, 0));
    LOADB(bb[1], BOFF(1, 0));

    for (int kk = 0; kk < EE; kk += 64) {
        // near-cache prefetch of the A stream ahead
        __builtin_prefetch(abase + kk + 512, 0, 3);
        // A for second half-step (one-ahead)
        a1.q[0] = *(const u32x4*)(abase + kk + 32);
        a1.q[1] = *(const u32x4*)(abase + kk + 48);
        GEMM_STEP(kk, a0);
        // A for next iteration's first half-step (final overrun stays inside ws)
        a0.q[0] = *(const u32x4*)(abase + kk + 64);
        a0.q[1] = *(const u32x4*)(abase + kk + 80);
        GEMM_STEP(kk + 32, a1);
    }

    // Epilogue: C layout -> row r+8*hh, col = nbase+t*16+col
    #pragma unroll
    for (int t = 0; t < 4; ++t) {
        #pragma unroll
        for (int r = 0; r < 8; ++r) {
            const int row = mrow + r + 8 * hh;
            const int n   = nbase + t * 16 + col;
            qo[(size_t)row * EE + n] = f32_to_bf16(acc[0 + t][r]);
            ko[(size_t)row * EE + n] = f32_to_bf16(acc[4 + t][r]);
            const int h = n >> 6, d = n & 63;          // V transposed: [d][g=h]
            vto[(size_t)row * EE + d * HH + h] = f32_to_bf16(acc[8 + t][r]);
        }
    }
}

// ---------------------------------------------------------------- attention
// One wave per token: scores = q k^T (16x16x64, 2 WMMAs), *pattern, softmax,
// out = attn v (K=16 zero-padded to 32, 4 WMMAs over D).
__global__ __launch_bounds__(256) void attn_kernel(
        const unsigned short* __restrict__ qb,
        const unsigned short* __restrict__ kb,
        const unsigned short* __restrict__ vtb,
        const float* __restrict__ pattern,
        float* __restrict__ out)
{
    __shared__ __align__(16) unsigned short attn_lds[8][16][16];

    const int lane  = threadIdx.x & 31;
    const int wave  = threadIdx.x >> 5;
    const int col   = lane & 15;
    const int hh    = lane >> 4;
    const int token = blockIdx.x * 8 + wave;

    const unsigned short* qrow = qb + (size_t)token * EE + (size_t)col * DD;
    const unsigned short* krow = kb + (size_t)token * EE + (size_t)col * DD;

    f32x8 sc = f32x8{0.f, 0.f, 0.f, 0.f, 0.f, 0.f, 0.f, 0.f};
    #pragma unroll
    for (int kk = 0; kk < DD; kk += 32) {
        FragAB a, b;
        a.q[0] = *(const u32x4*)(qrow + kk +      hh * 8);
        a.q[1] = *(const u32x4*)(qrow + kk + 16 + hh * 8);
        b.q[0] = *(const u32x4*)(krow + kk +      hh * 8);
        b.q[1] = *(const u32x4*)(krow + kk + 16 + hh * 8);
        sc = __builtin_amdgcn_wmma_f32_16x16x32_bf16(
            false, a.bf, false, b.bf, (short)0, sc, false, false);
    }

    // sparsity mask (pre-softmax multiply, per reference) + row softmax.
    // C layout: vector elem r = row (r+8*hh), lanes of each half = cols 0..15.
    #pragma unroll
    for (int r = 0; r < 8; ++r) {
        float s = sc[r] * pattern[(r + 8 * hh) * HH + col];
        float mx = s;
        mx = fmaxf(mx, __shfl_xor(mx, 1));
        mx = fmaxf(mx, __shfl_xor(mx, 2));
        mx = fmaxf(mx, __shfl_xor(mx, 4));
        mx = fmaxf(mx, __shfl_xor(mx, 8));
        float ex = __expf(s - mx);
        float sm = ex;
        sm += __shfl_xor(sm, 1);
        sm += __shfl_xor(sm, 2);
        sm += __shfl_xor(sm, 4);
        sm += __shfl_xor(sm, 8);
        float a = ex / sm;
        attn_lds[wave][r + 8 * hh][col] = f32_to_bf16(a);  // row-major for A re-read
    }
    __syncthreads();

    // attn as A fragment: lane row = col, K(g)=0..15 live in V0-3; K 16..31 pad = 0
    FragAB aa;
    aa.q[0] = *(const u32x4*)(&attn_lds[wave][col][hh * 8]);
    aa.q[1] = u32x4{0u, 0u, 0u, 0u};

    const unsigned short* vt = vtb + (size_t)token * EE;   // [d][g] per token
    #pragma unroll
    for (int t = 0; t < 4; ++t) {
        FragAB b;            // B col d = t*16+col, K=g 0..15 in V0-3 (8*hh+s); V4-7 pad zeros
        {
            const unsigned short* p = vt + (t * 16 + col) * HH + hh * 8;
            b.q[0] = *(const u32x4*)(p);
            b.q[1] = u32x4{0u, 0u, 0u, 0u};
        }
        f32x8 o = f32x8{0.f, 0.f, 0.f, 0.f, 0.f, 0.f, 0.f, 0.f};
        o = __builtin_amdgcn_wmma_f32_16x16x32_bf16(
            false, aa.bf, false, b.bf, (short)0, o, false, false);
        #pragma unroll
        for (int r = 0; r < 8; ++r)
            out[(size_t)token * EE + (r + 8 * hh) * DD + t * 16 + col] = o[r];
    }
}

// ---------------------------------------------------------------- launch
extern "C" void kernel_launch(void* const* d_in, const int* in_sizes, int n_in,
                              void* d_out, int out_size, void* d_ws, size_t ws_size,
                              hipStream_t stream) {
    const float* x       = (const float*)d_in[0];
    const float* pattern = (const float*)d_in[1];
    const float* Wq      = (const float*)d_in[2];
    const float* bq      = (const float*)d_in[3];
    const float* Wk      = (const float*)d_in[4];
    const float* bk      = (const float*)d_in[5];
    const float* Wv      = (const float*)d_in[6];
    const float* bv      = (const float*)d_in[7];
    float* out = (float*)d_out;

    // workspace layout (bf16 = unsigned short), total ~134 MB
    // wqb/wkb/wvb MUST stay contiguous (qkv_gemm_kernel indexes them as one).
    unsigned short* xb  = (unsigned short*)d_ws;
    unsigned short* wqb = xb  + (size_t)M_TOTAL * EE;
    unsigned short* wkb = wqb + (size_t)W_ELEMS;
    unsigned short* wvb = wkb + (size_t)W_ELEMS;
    unsigned short* qb  = wvb + (size_t)W_ELEMS;
    unsigned short* kb  = qb  + (size_t)M_TOTAL * EE;
    unsigned short* vtb = kb  + (size_t)M_TOTAL * EE;

    // fp32 -> bf16 conversions (vectorized x4)
    cvt_bf16_kernel<<<(M_TOTAL * EE / 4 + 255) / 256, 256, 0, stream>>>(x,  xb,  M_TOTAL * EE / 4);
    cvt_bf16_kernel<<<(W_ELEMS / 4 + 255) / 256,      256, 0, stream>>>(Wq, wqb, W_ELEMS / 4);
    cvt_bf16_kernel<<<(W_ELEMS / 4 + 255) / 256,      256, 0, stream>>>(Wk, wkb, W_ELEMS / 4);
    cvt_bf16_kernel<<<(W_ELEMS / 4 + 255) / 256,      256, 0, stream>>>(Wv, wvb, W_ELEMS / 4);

    // fused QKV projection: grid (16384/16 M-tiles, 1024/512 N-halves)
    qkv_gemm_kernel<<<dim3(M_TOTAL / 16, EE / 512), 256, 0, stream>>>(
        xb, wqb, bq, bk, bv, qb, kb, vtb);

    // per-token sparse attention: 8 tokens (waves) per block
    attn_kernel<<<M_TOTAL / 8, 256, 0, stream>>>(qb, kb, vtb, pattern, out);
}
